// GATFraudDetector_29772713296005
// MI455X (gfx1250) — matrix operand order repaired and checked by reading, not compiled, not run on hardware
//
#include <hip/hip_runtime.h>
#include <math.h>

// ---------------------------------------------------------------------------
// Problem constants (from the reference)
// ---------------------------------------------------------------------------
#define NA   100000   // accounts
#define NM   30000    // merchants
#define NE   200000   // edges per direction
#define HEADS 4

// ---------------------------------------------------------------------------
// WMMA / vector types (gfx1250, wave32)
// ---------------------------------------------------------------------------
typedef __bf16 bf16_t;
typedef bf16_t v16bf __attribute__((ext_vector_type(16)));
typedef float  v8f   __attribute__((ext_vector_type(8)));
typedef float  f4    __attribute__((ext_vector_type(4)));
typedef unsigned short us8 __attribute__((ext_vector_type(8)));
typedef unsigned short us4 __attribute__((ext_vector_type(4)));
typedef int    v4i   __attribute__((ext_vector_type(4)));
typedef __attribute__((address_space(1))) v4i* as1_v4i_ptr;
typedef __attribute__((address_space(3))) v4i* as3_v4i_ptr;

__device__ __forceinline__ unsigned short f2bf(float f) {
  unsigned int u = __float_as_uint(f);
  unsigned int r = u + 0x7FFFu + ((u >> 16) & 1u);   // round-to-nearest-even
  return (unsigned short)(r >> 16);
}

// ---------------------------------------------------------------------------
// gfx1250 async global->LDS copy (ASYNCcnt path), with synchronous fallback.
// Prototype discovered by probe: (v4i AS1*, v4i AS3*, imm offset, imm cpol).
// ---------------------------------------------------------------------------
#if defined(__has_builtin)
#if __has_builtin(__builtin_amdgcn_global_load_async_to_lds_b128) && \
    __has_builtin(__builtin_amdgcn_s_wait_asynccnt)
#define HAVE_ASYNC_LDS 1
#endif
#endif

__device__ __forceinline__ void copy16_g2l(const unsigned short* g, unsigned short* l) {
#ifdef HAVE_ASYNC_LDS
  as1_v4i_ptr gp = (as1_v4i_ptr)(unsigned long long)(const void*)g;
  as3_v4i_ptr lp = (as3_v4i_ptr)(unsigned int)(unsigned long long)(void*)l;
  __builtin_amdgcn_global_load_async_to_lds_b128(gp, lp, 0, 0);
#else
  *(us8*)l = *(const us8*)g;
#endif
}

__device__ __forceinline__ void wait_async_copies() {
#ifdef HAVE_ASYNC_LDS
  __builtin_amdgcn_s_wait_asynccnt(0);
#endif
}

// ---------------------------------------------------------------------------
// bf16 WMMA GEMM:  C[M,N] = A[M,K] * Bt[N,K]^T (+bias[N]) (optional ReLU)
//   - A: bf16 row-major [M,K]; Bt: bf16 pre-transposed weights [N,K]
//   - block tile 128x128, 8 waves, wave tile 32x64 = 2x4 WMMA tiles
//   - double-buffered LDS staged by global_load_async_to_lds_b128 (DMA
//     overlapped with the 8 v_wmma_f32_16x16x32_bf16 per K-step)
// Requirements: K % 32 == 0; M, N guarded (clamped loads, guarded stores).
// ---------------------------------------------------------------------------
#define TILE_M 128
#define TILE_N 128
#define TILE_K 32
#define LDA_S  40   // padded LDS K-stride (bf16 elems): 80B rows, 16B aligned
#define LDB_S  40

__global__ __launch_bounds__(256)
void wmma_gemm_bf16(const unsigned short* __restrict__ A,
                    const unsigned short* __restrict__ Bt,
                    const float* __restrict__ bias, float* __restrict__ C,
                    int M, int N, int K, int act) {
  __shared__ __attribute__((aligned(16))) unsigned short As[2][TILE_M * LDA_S]; // [m][k]
  __shared__ __attribute__((aligned(16))) unsigned short Bs[2][TILE_N * LDB_S]; // [n][k]

  const int tid  = threadIdx.x;
  const int lane = tid & 31;
  const int wave = tid >> 5;
  const int wm   = (wave & 3) * 32;   // 4 waves along M
  const int wn   = (wave >> 2) * 64;  // 2 waves along N
  const int rowBase = blockIdx.y * TILE_M;
  const int colBase = blockIdx.x * TILE_N;

  const int lhalf = lane >> 4;   // 0 | 1
  const int lmod  = lane & 15;

  v8f acc[2][4] = {};

  // stage one 128x32 + 128x32 bf16 tile pair into LDS buffer b (16B per copy)
  auto stage = [&](int b, int kk) {
#pragma unroll
    for (int i = 0; i < 2; ++i) {
      int idx = tid + i * 256;               // 512 x 16B chunks for A
      int r = idx >> 2, c16 = (idx & 3) * 8; // 4 chunks per 32-elem row
      int gr = rowBase + r; if (gr >= M) gr = M - 1;
      copy16_g2l(A + (size_t)gr * K + kk + c16, &As[b][r * LDA_S + c16]);
    }
#pragma unroll
    for (int i = 0; i < 2; ++i) {
      int idx = tid + i * 256;               // 512 x 16B chunks for Bt
      int r = idx >> 2, c16 = (idx & 3) * 8;
      int gn = colBase + r; if (gn >= N) gn = N - 1;
      copy16_g2l(Bt + (size_t)gn * K + kk + c16, &Bs[b][r * LDB_S + c16]);
    }
  };

  stage(0, 0);
  wait_async_copies();
  __syncthreads();

  int buf = 0;
  for (int k0 = 0; k0 < K; k0 += TILE_K) {
    // issue DMA for the next K-step before computing on this one
    if (k0 + TILE_K < K) {
      stage(buf ^ 1, k0 + TILE_K);
      if (tid < 128 && k0 + 2 * TILE_K < K) {   // GL2 prefetch two steps ahead
        int gr = rowBase + tid; if (gr >= M) gr = M - 1;
        __builtin_prefetch(A + (size_t)gr * K + k0 + 2 * TILE_K, 0, 1);
      }
    }

    // ---- 2x4 WMMA tiles per wave ----
#pragma unroll
    for (int tm = 0; tm < 2; ++tm) {
      // A fragment (ISA 7.12.2, 16-bit A 16x32):
      //   row = lane%16; lanes 0-15: K[0..7]+K[16..23]; lanes 16-31: K[8..15]+K[24..31]
      union { us8 h[2]; v16bf v; } af;
      const unsigned short* ap = &As[buf][(wm + tm * 16 + lmod) * LDA_S + lhalf * 8];
      af.h[0] = *(const us8*)(ap);
      af.h[1] = *(const us8*)(ap + 16);
#pragma unroll
      for (int tn = 0; tn < 4; ++tn) {
        // B fragment (32x16): N = lane%16; lanes 0-15: K[0..7]+K[16..23]; lanes 16-31: K[8..15]+K[24..31]
        union { us8 h[2]; v16bf v; } bfm;
        const unsigned short* bp = &Bs[buf][(wn + tn * 16 + lmod) * LDB_S + lhalf * 8];
        bfm.h[0] = *(const us8*)(bp);
        bfm.h[1] = *(const us8*)(bp + 16);
        acc[tm][tn] = __builtin_amdgcn_wmma_f32_16x16x32_bf16(
            false, af.v, false, bfm.v, (short)0, acc[tm][tn], false, false);
      }
    }

    wait_async_copies();   // next buffer's DMA complete (this wave)
    __syncthreads();       // all waves done reading buf & staging buf^1
    buf ^= 1;
  }

  // ---- store (C/D layout: lane%16 = N col; VGPR v = row v + 8*(lane/16)) ----
#pragma unroll
  for (int tm = 0; tm < 2; ++tm) {
#pragma unroll
    for (int tn = 0; tn < 4; ++tn) {
      int c = colBase + wn + tn * 16 + lmod;
      if (c < N) {
        float bv = bias ? bias[c] : 0.0f;
#pragma unroll
        for (int v = 0; v < 8; ++v) {
          int r = rowBase + wm + tm * 16 + v + 8 * lhalf;
          if (r < M) {
            float x = acc[tm][tn][v] + bv;
            if (act == 1) x = fmaxf(x, 0.0f);
            C[(size_t)r * N + c] = x;
          }
        }
      }
    }
  }
}

// ---------------------------------------------------------------------------
// fp32 -> bf16 conversion pre-passes
// ---------------------------------------------------------------------------
__global__ __launch_bounds__(256)
void cvt_bf16_k(const float* __restrict__ x, unsigned short* __restrict__ y, long n4) {
  long t = blockIdx.x * 256L + threadIdx.x;
  if (t >= n4) return;
  f4 v = *(const f4*)(x + t * 4);
  us4 h; h.x = f2bf(v.x); h.y = f2bf(v.y); h.z = f2bf(v.z); h.w = f2bf(v.w);
  *(us4*)(y + t * 4) = h;
}

// W[K,N] fp32 -> Bt[N,K] bf16 (weights are small: <= 512x512)
__global__ __launch_bounds__(256)
void cvt_t_bf16_k(const float* __restrict__ Wm, unsigned short* __restrict__ Bt,
                  int K, int N) {
  int t = blockIdx.x * 256 + threadIdx.x;
  if (t >= K * N) return;
  int k = t / N, n = t % N;
  Bt[(size_t)n * K + k] = f2bf(Wm[t]);
}

// ---------------------------------------------------------------------------
// Graph / elementwise kernels
// ---------------------------------------------------------------------------
__device__ __forceinline__ unsigned fkey(float f) {
  unsigned u = __float_as_uint(f);
  return (u & 0x80000000u) ? ~u : (u | 0x80000000u);
}
__device__ __forceinline__ float funkey(unsigned k) {
  unsigned u = (k & 0x80000000u) ? (k & 0x7FFFFFFFu) : ~k;
  return __uint_as_float(u);
}

__global__ __launch_bounds__(256)
void att_scores_k(const float* __restrict__ hbuf, const float* __restrict__ att,
                  float* __restrict__ out, int N, int C) {
  int t = blockIdx.x * 256 + threadIdx.x;
  if (t >= N * HEADS) return;
  int n = t >> 2, h = t & 3;
  const float* hp = hbuf + (size_t)n * (HEADS * C) + h * C;
  const float* ap = att + h * C;
  float s = 0.f;
  for (int c = 0; c < C; c += 4) {
    f4 hv = *(const f4*)(hp + c);
    f4 av = *(const f4*)(ap + c);
    s += hv.x * av.x + hv.y * av.y + hv.z * av.z + hv.w * av.w;
  }
  out[t] = s;
}

__global__ __launch_bounds__(256)
void init_mkey_k(unsigned* mkey, int n) {
  int t = blockIdx.x * 256 + threadIdx.x;
  if (t < n) mkey[t] = fkey(-__builtin_inff());
}

__global__ __launch_bounds__(256)
void edge_logits_k(const float* __restrict__ a_s, const float* __restrict__ a_d,
                   const int* __restrict__ src, const int* __restrict__ dst,
                   float* __restrict__ ebuf, unsigned* __restrict__ mkey, int total) {
  int t = blockIdx.x * 256 + threadIdx.x;
  if (t >= total) return;
  int e = t >> 2, h = t & 3;
  float v = a_s[src[e] * HEADS + h] + a_d[dst[e] * HEADS + h];
  v = (v > 0.f) ? v : 0.2f * v;
  ebuf[t] = v;
  atomicMax(&mkey[dst[e] * HEADS + h], fkey(v));
}

__global__ __launch_bounds__(256)
void finalize_m_k(unsigned* mkey, int n) {
  int t = blockIdx.x * 256 + threadIdx.x;
  if (t >= n) return;
  float m = funkey(mkey[t]);
  if (!isfinite(m)) m = 0.f;
  ((float*)mkey)[t] = m;
}

__global__ __launch_bounds__(256)
void edge_exp_k(const float* __restrict__ ebuf, const float* __restrict__ mf,
                const int* __restrict__ dst, float* __restrict__ alpha,
                float* __restrict__ den, int total) {
  int t = blockIdx.x * 256 + threadIdx.x;
  if (t >= total) return;
  int e = t >> 2, h = t & 3;
  int d = dst[e];
  float x = expf(ebuf[t] - mf[d * HEADS + h]);
  alpha[t] = x;
  atomicAdd(&den[d * HEADS + h], x);
}

__global__ __launch_bounds__(256)
void edge_norm_k(float* __restrict__ alpha, const float* __restrict__ den,
                 const int* __restrict__ dst, int total) {
  int t = blockIdx.x * 256 + threadIdx.x;
  if (t >= total) return;
  int e = t >> 2, h = t & 3;
  alpha[t] = alpha[t] / (den[dst[e] * HEADS + h] + 1e-16f);
}

// agg[dst, ch..ch+3] += h_src[src, ch..ch+3] * alpha[e,h]  (float4 gather)
__global__ __launch_bounds__(256)
void scatter_k(const float* __restrict__ hs, const float* __restrict__ alpha,
               const int* __restrict__ src, const int* __restrict__ dst,
               float* __restrict__ agg, long total4, int HC, int C) {
  long t = blockIdx.x * 256L + threadIdx.x;
  if (t >= total4) return;
  int  q4 = HC >> 2;
  int  ch = (int)(t % q4) * 4;
  long e  = t / q4;
  int  h  = ch / C;
  float a = alpha[e * HEADS + h];
  f4 hv = *(const f4*)(hs + (size_t)src[e] * HC + ch);
  float* ap = agg + (size_t)dst[e] * HC + ch;
  atomicAdd(ap + 0, hv.x * a);
  atomicAdd(ap + 1, hv.y * a);
  atomicAdd(ap + 2, hv.z * a);
  atomicAdd(ap + 3, hv.w * a);
}

__global__ __launch_bounds__(256)
void bias_add_k(const float* __restrict__ agg, const float* __restrict__ bias,
                float* __restrict__ out, long total, int D) {
  long t = blockIdx.x * 256L + threadIdx.x;
  if (t >= total) return;
  out[t] = agg[t] + bias[(int)(t % D)];
}

__global__ __launch_bounds__(256)
void mean_heads_k(const float* __restrict__ agg, const float* __restrict__ bias,
                  float* __restrict__ out, int N, int C) {
  int t = blockIdx.x * 256 + threadIdx.x;
  if (t >= N * C) return;
  int n = t / C, c = t % C;
  const float* a = agg + (size_t)n * (HEADS * C);
  out[t] = 0.25f * (a[c] + a[C + c] + a[2 * C + c] + a[3 * C + c]) + bias[c];
}

__global__ __launch_bounds__(256)
void bn_stats_k(const float* __restrict__ x, float* __restrict__ sum,
                float* __restrict__ sumsq, int N, int D) {
  int col = blockIdx.x * 256 + threadIdx.x;
  if (col >= D) return;
  float s = 0.f, q = 0.f;
  for (int r = blockIdx.y; r < N; r += gridDim.y) {
    float v = x[(size_t)r * D + col];
    s += v; q += v * v;
  }
  atomicAdd(&sum[col], s);
  atomicAdd(&sumsq[col], q);
}

__global__ __launch_bounds__(256)
void bn_norm_k(float* __restrict__ x, const float* __restrict__ sum,
               const float* __restrict__ sumsq, const float* __restrict__ gamma,
               const float* __restrict__ beta, int N, int D, int elu) {
  long t = blockIdx.x * 256L + threadIdx.x;
  if (t >= (long)N * D) return;
  int c = (int)(t % D);
  float mu  = sum[c] / (float)N;
  float var = sumsq[c] / (float)N - mu * mu;
  float v = (x[t] - mu) * rsqrtf(var + 1e-5f) * gamma[c] + beta[c];
  if (elu) v = (v > 0.f) ? v : (expf(v) - 1.f);
  x[t] = v;
}

__global__ __launch_bounds__(256)
void clf_out_k(const float* __restrict__ hid, const float* __restrict__ W2,
               const float* __restrict__ b2, float* __restrict__ out, int N) {
  int n = blockIdx.x * 256 + threadIdx.x;
  if (n >= N) return;
  float s = b2[0];
  const float* hp = hid + (size_t)n * 64;
#pragma unroll
  for (int c = 0; c < 64; c += 4) {
    f4 hv = *(const f4*)(hp + c);
    f4 wv = *(const f4*)(W2 + c);
    s += hv.x * wv.x + hv.y * wv.y + hv.z * wv.z + hv.w * wv.w;
  }
  out[n] = s;
}

// ---------------------------------------------------------------------------
// Host-side orchestration
// ---------------------------------------------------------------------------
static inline int cdiv(long a, long b) { return (int)((a + b - 1) / b); }

static void gemm(const unsigned short* A, const unsigned short* Bt, const float* bias,
                 float* C, int M, int N, int K, int act, hipStream_t s) {
  dim3 grid(cdiv(N, TILE_N), cdiv(M, TILE_M));
  wmma_gemm_bf16<<<grid, 256, 0, s>>>(A, Bt, bias, C, M, N, K, act);
}

struct GatScratch {
  float *hs, *hd, *a_s, *a_d, *ebuf, *alpha, *den, *agg;
  unsigned* mkey;
  unsigned short *bws, *bwd;   // bf16 transposed weight scratch
};

static void gat_conv_host(const unsigned short* bsrc, int Ns, const unsigned short* bdst,
                          int Nd, const int* ei, const float* Wsrc, const float* Wdst,
                          const float* atts, const float* attd, const float* bias,
                          int inW, int C, int concat, const GatScratch& S,
                          float* out, hipStream_t s) {
  const int HC = HEADS * C;
  cvt_t_bf16_k<<<cdiv((long)inW * HC, 256), 256, 0, s>>>(Wsrc, S.bws, inW, HC);
  cvt_t_bf16_k<<<cdiv((long)inW * HC, 256), 256, 0, s>>>(Wdst, S.bwd, inW, HC);
  gemm(bsrc, S.bws, nullptr, S.hs, Ns, HC, inW, 0, s);
  gemm(bdst, S.bwd, nullptr, S.hd, Nd, HC, inW, 0, s);
  att_scores_k<<<cdiv((long)Ns * HEADS, 256), 256, 0, s>>>(S.hs, atts, S.a_s, Ns, C);
  att_scores_k<<<cdiv((long)Nd * HEADS, 256), 256, 0, s>>>(S.hd, attd, S.a_d, Nd, C);

  init_mkey_k<<<cdiv((long)Nd * HEADS, 256), 256, 0, s>>>(S.mkey, Nd * HEADS);
  (void)hipMemsetAsync(S.den, 0, (size_t)Nd * HEADS * sizeof(float), s);
  (void)hipMemsetAsync(S.agg, 0, (size_t)Nd * HC * sizeof(float), s);

  const int* src = ei;
  const int* dst = ei + NE;
  const int EH = NE * HEADS;
  edge_logits_k<<<cdiv(EH, 256), 256, 0, s>>>(S.a_s, S.a_d, src, dst, S.ebuf, S.mkey, EH);
  finalize_m_k<<<cdiv((long)Nd * HEADS, 256), 256, 0, s>>>(S.mkey, Nd * HEADS);
  edge_exp_k<<<cdiv(EH, 256), 256, 0, s>>>(S.ebuf, (const float*)S.mkey, dst, S.alpha, S.den, EH);
  edge_norm_k<<<cdiv(EH, 256), 256, 0, s>>>(S.alpha, S.den, dst, EH);

  long total4 = (long)NE * (HC >> 2);
  scatter_k<<<cdiv(total4, 256), 256, 0, s>>>(S.hs, S.alpha, src, dst, S.agg, total4, HC, C);

  if (concat) {
    long t2 = (long)Nd * HC;
    bias_add_k<<<cdiv(t2, 256), 256, 0, s>>>(S.agg, bias, out, t2, HC);
  } else {
    mean_heads_k<<<cdiv((long)Nd * C, 256), 256, 0, s>>>(S.agg, bias, out, Nd, C);
  }
}

static void bn_host(float* x, const float* gamma, const float* beta, int N, int D,
                    int elu, float* sum, float* sq, hipStream_t s) {
  (void)hipMemsetAsync(sum, 0, D * sizeof(float), s);
  (void)hipMemsetAsync(sq, 0, D * sizeof(float), s);
  dim3 g(cdiv(D, 256), 256);
  bn_stats_k<<<g, 256, 0, s>>>(x, sum, sq, N, D);
  long t = (long)N * D;
  bn_norm_k<<<cdiv(t, 256), 256, 0, s>>>(x, sum, sq, gamma, beta, N, D, elu);
}

extern "C" void kernel_launch(void* const* d_in, const int* in_sizes, int n_in,
                              void* d_out, int out_size, void* d_ws, size_t ws_size,
                              hipStream_t stream) {
  // -------- inputs, flattened in setup_inputs() dict insertion order --------
  int ix = 0;
  const float* x_account  = (const float*)d_in[ix++]; // [NA,64]
  const float* x_merchant = (const float*)d_in[ix++]; // [NM,32]
  const int*   ei_am      = (const int*)d_in[ix++];   // [2,E]
  const int*   ei_ma      = (const int*)d_in[ix++];   // [2,E]
  const float* pa_W = (const float*)d_in[ix++];       // proj.account.W [64,128]
  const float* pa_b = (const float*)d_in[ix++];
  const float* pm_W = (const float*)d_in[ix++];       // proj.merchant.W [32,128]
  const float* pm_b = (const float*)d_in[ix++];

  struct GatP { const float *Wsrc, *Wdst, *atts, *attd, *bias; };
  struct LayerP { GatP am, ma; const float *g_a, *b_a, *g_m, *b_m; };
  LayerP L[3];
  for (int i = 0; i < 3; ++i) {
    L[i].am.Wsrc = (const float*)d_in[ix++]; L[i].am.Wdst = (const float*)d_in[ix++];
    L[i].am.atts = (const float*)d_in[ix++]; L[i].am.attd = (const float*)d_in[ix++];
    L[i].am.bias = (const float*)d_in[ix++];
    L[i].ma.Wsrc = (const float*)d_in[ix++]; L[i].ma.Wdst = (const float*)d_in[ix++];
    L[i].ma.atts = (const float*)d_in[ix++]; L[i].ma.attd = (const float*)d_in[ix++];
    L[i].ma.bias = (const float*)d_in[ix++];
    L[i].g_a = (const float*)d_in[ix++]; L[i].b_a = (const float*)d_in[ix++]; // bn.account
    L[i].g_m = (const float*)d_in[ix++]; L[i].b_m = (const float*)d_in[ix++]; // bn.merchant
  }
  const float* cW1 = (const float*)d_in[ix++]; // [64,64]
  const float* cb1 = (const float*)d_in[ix++];
  const float* cW2 = (const float*)d_in[ix++]; // [64,1]
  const float* cb2 = (const float*)d_in[ix++];

  // -------- workspace carve-out --------
  char* p = (char*)d_ws;
  auto alloc = [&](size_t bytes) -> void* {
    void* r = (void*)p;
    p += (bytes + 255) & ~(size_t)255;
    return r;
  };
  const size_t W = 512; // max feature width
  float* xa0 = (float*)alloc((size_t)NA * W * 4);
  float* xa1 = (float*)alloc((size_t)NA * W * 4);
  float* xm0 = (float*)alloc((size_t)NM * W * 4);
  float* xm1 = (float*)alloc((size_t)NM * W * 4);
  unsigned short* bxa = (unsigned short*)alloc((size_t)NA * W * 2); // bf16 activations
  unsigned short* bxm = (unsigned short*)alloc((size_t)NM * W * 2);
  GatScratch S;
  S.hs    = (float*)alloc((size_t)NA * W * 4);
  S.hd    = (float*)alloc((size_t)NA * W * 4);
  S.agg   = (float*)alloc((size_t)NA * W * 4);
  S.a_s   = (float*)alloc((size_t)NA * HEADS * 4);
  S.a_d   = (float*)alloc((size_t)NA * HEADS * 4);
  S.ebuf  = (float*)alloc((size_t)NE * HEADS * 4);
  S.alpha = (float*)alloc((size_t)NE * HEADS * 4);
  S.mkey  = (unsigned*)alloc((size_t)NA * HEADS * 4);
  S.den   = (float*)alloc((size_t)NA * HEADS * 4);
  S.bws   = (unsigned short*)alloc((size_t)W * W * 2);
  S.bwd   = (unsigned short*)alloc((size_t)W * W * 2);
  float* bnsum = (float*)alloc(W * 4);
  float* bnsq  = (float*)alloc(W * 4);

  // -------- input projections (bf16 WMMA GEMMs) --------
  cvt_bf16_k<<<cdiv((long)NA * 64 / 4, 256), 256, 0, stream>>>(x_account, bxa, (long)NA * 64 / 4);
  cvt_t_bf16_k<<<cdiv(64 * 128, 256), 256, 0, stream>>>(pa_W, S.bws, 64, 128);
  gemm(bxa, S.bws, pa_b, xa0, NA, 128, 64, 0, stream);
  cvt_bf16_k<<<cdiv((long)NM * 32 / 4, 256), 256, 0, stream>>>(x_merchant, bxm, (long)NM * 32 / 4);
  cvt_t_bf16_k<<<cdiv(32 * 128, 256), 256, 0, stream>>>(pm_W, S.bws, 32, 128);
  gemm(bxm, S.bws, pm_b, xm0, NM, 128, 32, 0, stream);

  const int OUT_DIMS[3] = {128, 128, 64};
  const int CONCAT[3]   = {1, 1, 0};

  float* cxa = xa0; float* cxm = xm0;
  int inW = 128;
  for (int i = 0; i < 3; ++i) {
    int C = OUT_DIMS[i];
    int D = CONCAT[i] ? HEADS * C : C;
    float* nxa = (cxa == xa0) ? xa1 : xa0;
    float* nxm = (cxm == xm0) ? xm1 : xm0;

    // one bf16 conversion of each node table per layer, shared by both convs
    cvt_bf16_k<<<cdiv((long)NA * inW / 4, 256), 256, 0, stream>>>(cxa, bxa, (long)NA * inW / 4);
    cvt_bf16_k<<<cdiv((long)NM * inW / 4, 256), 256, 0, stream>>>(cxm, bxm, (long)NM * inW / 4);

    // account -> merchant conv (out_m)
    gat_conv_host(bxa, NA, bxm, NM, ei_am, L[i].am.Wsrc, L[i].am.Wdst,
                  L[i].am.atts, L[i].am.attd, L[i].am.bias, inW, C, CONCAT[i],
                  S, nxm, stream);
    // merchant -> account conv (out_a)
    gat_conv_host(bxm, NM, bxa, NA, ei_ma, L[i].ma.Wsrc, L[i].ma.Wdst,
                  L[i].ma.atts, L[i].ma.attd, L[i].ma.bias, inW, C, CONCAT[i],
                  S, nxa, stream);

    int elu = (i < 2) ? 1 : 0;
    bn_host(nxa, L[i].g_a, L[i].b_a, NA, D, elu, bnsum, bnsq, stream);
    bn_host(nxm, L[i].g_m, L[i].b_m, NM, D, elu, bnsum, bnsq, stream);

    cxa = nxa; cxm = nxm; inW = D;
  }

  // -------- classifier: relu(emb @ W1 + b1) @ W2 + b2 --------
  cvt_bf16_k<<<cdiv((long)NA * 64 / 4, 256), 256, 0, stream>>>(cxa, bxa, (long)NA * 64 / 4);
  cvt_t_bf16_k<<<cdiv(64 * 64, 256), 256, 0, stream>>>(cW1, S.bws, 64, 64);
  gemm(bxa, S.bws, cb1, S.hs, NA, 64, 64, 1, stream);           // WMMA + fused ReLU
  clf_out_k<<<cdiv(NA, 256), 256, 0, stream>>>(S.hs, cW2, cb2, (float*)d_out, NA);
}
